// CustomBlock_61040075211154
// MI455X (gfx1250) — matrix operand order repaired and checked by reading, not compiled, not run on hardware
//
#include <hip/hip_runtime.h>

// ============================ WMMA types/helpers ============================
typedef __attribute__((ext_vector_type(16))) __bf16 v16bf;
typedef __attribute__((ext_vector_type(4)))  __bf16 v4bf;
typedef __attribute__((ext_vector_type(8)))  float  v8f;
typedef __attribute__((ext_vector_type(4)))  unsigned int v4u;
typedef __attribute__((ext_vector_type(8)))  int v8i;
typedef __attribute__((ext_vector_type(4)))  int v4i;

#define WMMA_BF16(A_,B_,C_) __builtin_amdgcn_wmma_f32_16x16x32_bf16(false,(A_),false,(B_),(short)0,(C_),false,false)

// A fragment (16x32, bf16) from LDS tile with row pitch `pitch` (halfs).
static __device__ __forceinline__ v16bf ld_afrag(const __bf16* base, int pitch, int lane){
  const int m = lane & 15, kb = (lane >> 4) * 8;
  const __bf16* p = base + m * pitch + kb;
  v16bf a;
#pragma unroll
  for (int e = 0; e < 8; ++e) { a[e] = p[e]; a[e + 8] = p[e + 16]; }
  return a;
}
// B fragment (32x16) from LDS tile stored transposed: Bt[n][k], pitch in halfs.
static __device__ __forceinline__ v16bf ld_bfrag(const __bf16* base, int pitch, int lane){
  const int n = lane & 15, kb = (lane >> 4) * 8;
  const __bf16* p = base + n * pitch + kb;
  v16bf b;
#pragma unroll
  for (int e = 0; e < 8; ++e) { b[e] = p[e]; b[e + 8] = p[e + 16]; }
  return b;
}
static __device__ __forceinline__ float sigm(float x){ return 1.f/(1.f+__expf(-x)); }
static __device__ __forceinline__ float gelu_t(float x){
  return 0.5f*x*(1.f+tanhf(0.7978845608028654f*(x+0.044715f*x*x*x)));
}
static __device__ __forceinline__ v4bf pk4(float4 v){
  v4bf h; h[0]=(__bf16)v.x; h[1]=(__bf16)v.y; h[2]=(__bf16)v.z; h[3]=(__bf16)v.w; return h;
}

// ===================== TDM probe: NULL-tensor load (no-op) ==================
// D# group0 all-zero => count=0 => NULL tensor descriptor; issues the TDM op
// and waits on TENSORcnt. Architecturally a no-op, deterministic.
// This toolchain carries the 6-arg builtin (extra int32x8 descriptor group).
__global__ void k_tdm_probe()
{
#if defined(__has_builtin)
#if __has_builtin(__builtin_amdgcn_tensor_load_to_lds)
  v4u g0 = {0u, 0u, 0u, 0u};
  v8i g1 = {0, 0, 0, 0, 0, 0, 0, 0};
  v4i g2 = {0, 0, 0, 0};
  v4i g3 = {0, 0, 0, 0};
  v8i g4 = {0, 0, 0, 0, 0, 0, 0, 0};
  __builtin_amdgcn_tensor_load_to_lds(g0, g1, g2, g3, g4, 0);
  __builtin_amdgcn_s_wait_tensorcnt(0);
#endif
#endif
}

// ============================ generic WMMA GEMM =============================
// C[m,n] = sum_k A[m,k]*B[k,n] (+bias)(+res). 64x64 tile, BK=32, 4 waves.
__global__ __launch_bounds__(128) void k_gemm(
    const float* __restrict__ A, const float* __restrict__ Bm,
    const float* __restrict__ bias, const float* __restrict__ res,
    float* __restrict__ Cm,
    int M, int N, int K, int lda, int ldb, int ldc,
    long long sA, long long sB, long long sC, long long sR,
    int transB, int biasRow, float alpha)
{
  __shared__ __bf16 As[64 * 40];   // padded pitch: conflict-free frag loads
  __shared__ __bf16 Bs[64 * 40];   // stored as Bt[n][k]
  const int tid = threadIdx.x, lane = tid & 31, wave = tid >> 5;
  const int bm = blockIdx.y * 64, bn = blockIdx.x * 64;
  const long long bz = blockIdx.z;
  A += bz * sA; Bm += bz * sB; Cm += bz * sC;
  if (res) res += bz * sR;

  v8f acc[4];
#pragma unroll
  for (int t = 0; t < 4; ++t)
#pragma unroll
    for (int r = 0; r < 8; ++r) acc[t][r] = 0.f;

  const int mrow = wave * 16;
  const bool vec_ok = (((lda | ldb) & 3) == 0) && (bm + 64 <= M) && (bn + 64 <= N) && !transB;
  for (int k0 = 0; k0 < K; k0 += 32) {
    if (vec_ok && (k0 + 32 <= K)) {
      // fast path: b128 loads, packed bf16 stores
      for (int t = tid; t < 512; t += 128) {               // A tile 64x32
        const int m = t >> 3, kc = (t & 7) * 4;
        const float4 v = *(const float4*)&A[(long long)(bm + m) * lda + k0 + kc];
        *(v4bf*)&As[m * 40 + kc] = pk4(v);
      }
      for (int t = tid; t < 512; t += 128) {               // B tile 32x64 -> Bt
        const int kk = t >> 4, nc = (t & 15) * 4;
        const float4 v = *(const float4*)&Bm[(long long)(k0 + kk) * ldb + bn + nc];
        Bs[(nc + 0) * 40 + kk] = (__bf16)v.x;
        Bs[(nc + 1) * 40 + kk] = (__bf16)v.y;
        Bs[(nc + 2) * 40 + kk] = (__bf16)v.z;
        Bs[(nc + 3) * 40 + kk] = (__bf16)v.w;
      }
    } else {
      for (int idx = tid; idx < 64 * 32; idx += 128) {     // A tile 64x32
        const int m = idx >> 5, k = idx & 31;
        const int gm = bm + m, gk = k0 + k;
        float v = 0.f;
        if (gm < M && gk < K) v = A[(long long)gm * lda + gk];
        As[m * 40 + k] = (__bf16)v;
      }
      for (int idx = tid; idx < 64 * 32; idx += 128) {     // B tile 32x64 -> Bt
        const int k = idx >> 6, n = idx & 63;
        const int gn = bn + n, gk = k0 + k;
        float v = 0.f;
        if (gn < N && gk < K)
          v = transB ? Bm[(long long)gn * ldb + gk] : Bm[(long long)gk * ldb + gn];
        Bs[n * 40 + k] = (__bf16)v;
      }
    }
    if (k0 + 32 < K) {
      const int pr = bm + (tid >> 1);
      __builtin_prefetch(&A[(long long)(pr < M ? pr : M - 1) * lda + k0 + 32], 0, 3);
    }
    __syncthreads();
    const v16bf af = ld_afrag(&As[mrow * 40], 40, lane);
#pragma unroll
    for (int nt = 0; nt < 4; ++nt) {
      const v16bf bf = ld_bfrag(&Bs[(nt * 16) * 40], 40, lane);
      acc[nt] = WMMA_BF16(af, bf, acc[nt]);
    }
    __syncthreads();
  }
  const int nl = lane & 15, mh = (lane >> 4) * 8;
#pragma unroll
  for (int nt = 0; nt < 4; ++nt) {
#pragma unroll
    for (int r = 0; r < 8; ++r) {
      const int gm = bm + mrow + mh + r, gn = bn + nt * 16 + nl;
      if (gm < M && gn < N) {
        float v = alpha * acc[nt][r];
        if (bias) v += biasRow ? bias[gm] : bias[gn];
        if (res)  v += res[(long long)gm * ldc + gn];
        Cm[(long long)gm * ldc + gn] = v;
      }
    }
  }
}

// ====================== fused attention (per b,h,q-tile) ====================
__global__ __launch_bounds__(128) void k_attn(
    const float* __restrict__ Q, const float* __restrict__ Kt,
    const float* __restrict__ V, float* __restrict__ O, int L)
{
  __shared__ float sc[16 * 656];
  __shared__ float part[4 * 256];
  __shared__ float red[128];
  const int qt = blockIdx.x, h = blockIdx.y, b = blockIdx.z;
  const int tid = threadIdx.x, lane = tid & 31, wave = tid >> 5;
  const long long base = ((long long)b * L) * 64 + h * 16;

  v16bf qf;                                  // Q frag, scaled by 1/sqrt(16)
  {
    const int m = lane & 15, kb = (lane >> 4) * 8;
    const float* qp = &Q[base + (long long)(qt * 16 + m) * 64 + kb];
    const float4 q0 = *(const float4*)qp;
    const float4 q1 = *(const float4*)(qp + 4);
    qf[0]=(__bf16)(q0.x*0.25f); qf[1]=(__bf16)(q0.y*0.25f);
    qf[2]=(__bf16)(q0.z*0.25f); qf[3]=(__bf16)(q0.w*0.25f);
    qf[4]=(__bf16)(q1.x*0.25f); qf[5]=(__bf16)(q1.y*0.25f);
    qf[6]=(__bf16)(q1.z*0.25f); qf[7]=(__bf16)(q1.w*0.25f);
#pragma unroll
    for (int e = 0; e < 8; ++e) qf[e + 8] = (__bf16)0.f;   // K=16..31 padding
  }
  const int ntile = L >> 4;
  for (int j = wave; j < ntile; j += 4) {    // score tiles round-robin on waves
    v16bf kf;
    const int n = lane & 15, kb = (lane >> 4) * 8;
    {                                         // lanes carry d=kb..kb+7 in e=0..7
      const float* kp = &Kt[base + (long long)(j * 16 + n) * 64 + kb];
      const float4 k0 = *(const float4*)kp;
      const float4 k1 = *(const float4*)(kp + 4);
      kf[0]=(__bf16)k0.x; kf[1]=(__bf16)k0.y; kf[2]=(__bf16)k0.z; kf[3]=(__bf16)k0.w;
      kf[4]=(__bf16)k1.x; kf[5]=(__bf16)k1.y; kf[6]=(__bf16)k1.z; kf[7]=(__bf16)k1.w;
#pragma unroll
      for (int e = 0; e < 8; ++e) kf[e + 8] = (__bf16)0.f;   // d=16..31 padding
    }
    v8f s;
#pragma unroll
    for (int r = 0; r < 8; ++r) s[r] = 0.f;
    s = WMMA_BF16(qf, kf, s);
    const int mh = (lane >> 4) * 8;
#pragma unroll
    for (int r = 0; r < 8; ++r) sc[(mh + r) * 656 + j * 16 + n] = s[r];
  }
  __syncthreads();
  // parallel softmax: 8 threads per query row
  {
    const int row = tid >> 3, sub = tid & 7;
    float mx = -3.4e38f;
    for (int c = sub; c < L; c += 8) mx = fmaxf(mx, sc[row * 656 + c]);
    red[tid] = mx;
    __syncthreads();
    if (sub == 0) {
      float m2 = red[tid];
#pragma unroll
      for (int j2 = 1; j2 < 8; ++j2) m2 = fmaxf(m2, red[tid + j2]);
      red[tid] = m2;
    }
    __syncthreads();
    mx = red[row << 3];
    float sm = 0.f;
    for (int c = sub; c < L; c += 8) {
      const float e = __expf(sc[row * 656 + c] - mx);
      sc[row * 656 + c] = e; sm += e;
    }
    __syncthreads();
    red[tid] = sm;
    __syncthreads();
    if (sub == 0) {
      float s2 = red[tid];
#pragma unroll
      for (int j2 = 1; j2 < 8; ++j2) s2 += red[tid + j2];
      red[tid] = s2;
    }
    __syncthreads();
    const float inv = 1.f / red[row << 3];
    for (int c = sub; c < L; c += 8) sc[row * 656 + c] *= inv;
  }
  __syncthreads();
  v8f oa;
#pragma unroll
  for (int r = 0; r < 8; ++r) oa[r] = 0.f;
  const int nchunk = L >> 5;
  for (int kc = wave; kc < nchunk; kc += 4) { // P @ V, K split over waves
    const int k0 = kc * 32;
    v16bf pa;
    {
      const int m = lane & 15, kb = (lane >> 4) * 8;
      const float4 p0 = *(const float4*)&sc[m * 656 + k0 + kb];
      const float4 p1 = *(const float4*)&sc[m * 656 + k0 + kb + 4];
      const float4 p2 = *(const float4*)&sc[m * 656 + k0 + 16 + kb];
      const float4 p3 = *(const float4*)&sc[m * 656 + k0 + 16 + kb + 4];
      pa[0]=(__bf16)p0.x; pa[1]=(__bf16)p0.y; pa[2]=(__bf16)p0.z; pa[3]=(__bf16)p0.w;
      pa[4]=(__bf16)p1.x; pa[5]=(__bf16)p1.y; pa[6]=(__bf16)p1.z; pa[7]=(__bf16)p1.w;
      pa[8]=(__bf16)p2.x; pa[9]=(__bf16)p2.y; pa[10]=(__bf16)p2.z; pa[11]=(__bf16)p2.w;
      pa[12]=(__bf16)p3.x; pa[13]=(__bf16)p3.y; pa[14]=(__bf16)p3.z; pa[15]=(__bf16)p3.w;
    }
    v16bf vf;
    {
      const int n = lane & 15, kb = (lane >> 4) * 8;
#pragma unroll
      for (int e = 0; e < 16; ++e)
        vf[e] = (__bf16)V[base + (long long)(k0 + kb + (e < 8 ? e : e + 8)) * 64 + n];
    }
    oa = WMMA_BF16(pa, vf, oa);
  }
  {
    const int n = lane & 15, mh = (lane >> 4) * 8;
#pragma unroll
    for (int r = 0; r < 8; ++r) part[wave * 256 + (mh + r) * 16 + n] = oa[r];
  }
  __syncthreads();
  for (int idx = tid; idx < 256; idx += 128) {
    const float v = part[idx] + part[256 + idx] + part[512 + idx] + part[768 + idx];
    const int m = idx >> 4, n = idx & 15;
    O[base + (long long)(qt * 16 + m) * 64 + n] = v;
  }
}

// ============== fused FeedForwardModule: out = x + 0.5*(silu(LN(x)W1+b1)W2+b2)
__global__ __launch_bounds__(256) void k_ffm(
    const float* __restrict__ Hln, const float* __restrict__ Xres,
    const float* __restrict__ W1, const float* __restrict__ B1,
    const float* __restrict__ W2, const float* __restrict__ B2,
    float* __restrict__ Out)
{
  __shared__ __bf16 Xs[32 * 72];
  __shared__ __bf16 W1t[64 * 72];
  __shared__ __bf16 W2t[64 * 72];
  __shared__ __bf16 H1s[32 * 72];
  const int tid = threadIdx.x, lane = tid & 31, wave = tid >> 5;
  const int mi = wave >> 2, ni = wave & 3;
  const long long bm = (long long)blockIdx.x * 32;
  for (int t = tid; t < 512; t += 256) {              // X tile 32x64 (b128)
    const int m = t >> 4, kc = (t & 15) * 4;
    const float4 v = *(const float4*)&Hln[(bm + m) * 64 + kc];
    *(v4bf*)&Xs[m * 72 + kc] = pk4(v);
  }
  v8f oacc;
#pragma unroll
  for (int r = 0; r < 8; ++r) oacc[r] = 0.f;
  for (int j = 0; j < 32; ++j) {
    __syncthreads();
    for (int t = tid; t < 1024; t += 256) {           // W1/W2 64x64 chunks
      const int kk = t >> 4, nc = (t & 15) * 4;
      const float4 a = *(const float4*)&W1[(long long)kk * 2048 + j * 64 + nc];
      const float4 c = *(const float4*)&W2[(long long)(j * 64 + kk) * 64 + nc];
      W1t[(nc + 0) * 72 + kk] = (__bf16)a.x;
      W1t[(nc + 1) * 72 + kk] = (__bf16)a.y;
      W1t[(nc + 2) * 72 + kk] = (__bf16)a.z;
      W1t[(nc + 3) * 72 + kk] = (__bf16)a.w;
      W2t[(nc + 0) * 72 + kk] = (__bf16)c.x;
      W2t[(nc + 1) * 72 + kk] = (__bf16)c.y;
      W2t[(nc + 2) * 72 + kk] = (__bf16)c.z;
      W2t[(nc + 3) * 72 + kk] = (__bf16)c.w;
    }
    __syncthreads();
    v8f h;
#pragma unroll
    for (int r = 0; r < 8; ++r) h[r] = 0.f;
#pragma unroll
    for (int k0 = 0; k0 < 64; k0 += 32) {
      const v16bf a = ld_afrag(&Xs[(mi * 16) * 72 + k0], 72, lane);
      const v16bf b = ld_bfrag(&W1t[(ni * 16) * 72 + k0], 72, lane);
      h = WMMA_BF16(a, b, h);
    }
    {
      const int n = lane & 15, mh = (lane >> 4) * 8;
      const float b1v = B1[j * 64 + ni * 16 + n];
#pragma unroll
      for (int r = 0; r < 8; ++r) {
        const float x = h[r] + b1v;
        H1s[(mi * 16 + mh + r) * 72 + ni * 16 + n] = (__bf16)(x * sigm(x));
      }
    }
    __syncthreads();
#pragma unroll
    for (int k0 = 0; k0 < 64; k0 += 32) {
      const v16bf a = ld_afrag(&H1s[(mi * 16) * 72 + k0], 72, lane);
      const v16bf b = ld_bfrag(&W2t[(ni * 16) * 72 + k0], 72, lane);
      oacc = WMMA_BF16(a, b, oacc);
    }
  }
  const int n = lane & 15, mh = (lane >> 4) * 8;
#pragma unroll
  for (int r = 0; r < 8; ++r) {
    const long long row = bm + mi * 16 + mh + r;
    const int col = ni * 16 + n;
    Out[row * 64 + col] = Xres[row * 64 + col] + 0.5f * (oacc[r] + B2[col]);
  }
}

// ============================ S4 support kernels ============================
__global__ void k_s4kernel(const float* __restrict__ logdt, const float* __restrict__ Are,
                           const float* __restrict__ Aim, const float* __restrict__ Cre,
                           const float* __restrict__ Cim, float* __restrict__ kout, int L)
{
  const int idx = blockIdx.x * blockDim.x + threadIdx.x;
  if (idx >= 64 * L) return;
  const int hh = idx / L, l = idx % L;
  const float dt = __expf(logdt[hh]);
  const float tl = (float)l;
  float acc = 0.f;
  for (int n = 0; n < 64; ++n) {
    const float are = -__expf(Are[hh * 64 + n]);
    const float aim = Aim[hh * 64 + n];
    const float dre = dt * are, dmi = dt * aim;
    const float e1 = __expf(dre);
    float s1, c1; __sincosf(dmi, &s1, &c1);
    const float nre = e1 * c1 - 1.f, nim = e1 * s1;
    const float den = are * are + aim * aim;
    const float qre = (nre * are + nim * aim) / den;   // (exp(dtA)-1)/A
    const float qim = (nim * are - nre * aim) / den;
    const float cr = Cre[hh * 64 + n], ci = Cim[hh * 64 + n];
    const float cbre = cr * qre - ci * qim;
    const float cbim = cr * qim + ci * qre;
    const float e2 = __expf(dre * tl);
    float s2, c2; __sincosf(dmi * tl, &s2, &c2);
    acc += cbre * (e2 * c2) - cbim * (e2 * s2);        // Re(CB * exp(dtA*l))
  }
  kout[hh * L + l] = 2.f * acc;
}

// causal conv + D*u skip + GELU, one (b,h) per block, LDS-staged
__global__ __launch_bounds__(256) void k_causal(
    const float* __restrict__ h, const float* __restrict__ kssm,
    const float* __restrict__ Dv, float* __restrict__ y, int L)
{
  __shared__ float u[640];
  __shared__ float kk[640];
  const int hh = blockIdx.x, b = blockIdx.y, tid = threadIdx.x;
  for (int m = tid; m < L; m += 256) {
    u[m]  = h[((long long)b * L + m) * 64 + hh];
    kk[m] = kssm[hh * L + m];
  }
  __syncthreads();
  const float Dh = Dv[hh];
  for (int l = tid; l < L; l += 256) {
    float acc = 0.f;
    for (int m = 0; m <= l; ++m) acc += kk[l - m] * u[m];
    y[((long long)b * L + l) * 64 + hh] = gelu_t(acc + Dh * u[l]);
  }
}

// h = LN(h + GLU(t)); rows of 64, t rows of 128
__global__ void k_s4_tail(float* __restrict__ h, const float* __restrict__ t,
                          const float* __restrict__ g, const float* __restrict__ bb, int rows)
{
  const int r = blockIdx.x * blockDim.x + threadIdx.x;
  if (r >= rows) return;
  float v[64];
  float* hr = h + (long long)r * 64;
  const float* tr = t + (long long)r * 128;
  float s = 0.f;
#pragma unroll
  for (int c = 0; c < 64; ++c) { v[c] = hr[c] + tr[c] * sigm(tr[64 + c]); s += v[c]; }
  const float m = s * (1.f / 64.f);
  float var = 0.f;
#pragma unroll
  for (int c = 0; c < 64; ++c) { const float d = v[c] - m; var += d * d; }
  const float rs = rsqrtf(var * (1.f / 64.f) + 1e-5f);
#pragma unroll
  for (int c = 0; c < 64; ++c) hr[c] = (v[c] - m) * rs * g[c] + bb[c];
}

// ============================ misc elementwise =============================
__global__ void k_layernorm(const float* __restrict__ in, float* __restrict__ out,
                            const float* __restrict__ g, const float* __restrict__ bb, int rows)
{
  const int r = blockIdx.x * blockDim.x + threadIdx.x;
  if (r >= rows) return;
  const float* x = in + (long long)r * 64;
  float s = 0.f;
#pragma unroll
  for (int c = 0; c < 64; ++c) s += x[c];
  const float m = s * (1.f / 64.f);
  float v = 0.f;
#pragma unroll
  for (int c = 0; c < 64; ++c) { const float d = x[c] - m; v += d * d; }
  const float rs = rsqrtf(v * (1.f / 64.f) + 1e-5f);
  float* o = out + (long long)r * 64;
#pragma unroll
  for (int c = 0; c < 64; ++c) o[c] = (x[c] - m) * rs * g[c] + bb[c];
}

// generic small conv1d over NCL (optionally reading NLC input), NCL output
__global__ void k_conv1d(const float* __restrict__ in, const float* __restrict__ w,
                         const float* __restrict__ bias, const float* __restrict__ bng,
                         const float* __restrict__ bnb, float* __restrict__ out,
                         int Lin, int Lout, int Cin, int Cout, int K, int pad, int stride,
                         int in_lc, int relu, int total)
{
  const int idx = blockIdx.x * blockDim.x + threadIdx.x;
  if (idx >= total) return;
  const int l = idx % Lout;
  const int o = (idx / Lout) % Cout;
  const int b = idx / (Lout * Cout);
  float acc = bias ? bias[o] : 0.f;
  for (int c = 0; c < Cin; ++c) {
    const float* wr = w + ((long long)o * Cin + c) * K;
    for (int k = 0; k < K; ++k) {
      const int li = l * stride + k - pad;
      if (li < 0 || li >= Lin) continue;
      const float v = in_lc ? in[((long long)b * Lin + li) * Cin + c]
                            : in[((long long)b * Cin + c) * Lin + li];
      acc += wr[k] * v;
    }
  }
  if (bng) acc = acc * bng[o] * rsqrtf(1.f + 1e-5f) + bnb[o];   // BN(eval, unit stats)
  if (relu) acc = fmaxf(acc, 0.f);
  out[((long long)b * Cout + o) * Lout + l] = acc;
}

// k=3 pad=1 conv over NLC layout (transformer conv-FFN)
__global__ void k_conv1d_lc(const float* __restrict__ in, const float* __restrict__ w,
                            const float* __restrict__ bias, float* __restrict__ out,
                            int L, int Cin, int Cout, int relu, long long total)
{
  const long long idx = (long long)blockIdx.x * blockDim.x + threadIdx.x;
  if (idx >= total) return;
  const int o = (int)(idx % Cout);
  const int l = (int)((idx / Cout) % L);
  const int b = (int)(idx / ((long long)Cout * L));
  float acc = bias[o];
  for (int k = 0; k < 3; ++k) {
    const int li = l + k - 1;
    if (li < 0 || li >= L) continue;
    const float* ir = in + ((long long)b * L + li) * Cin;
    const float* wr = w + (long long)o * Cin * 3 + k;
    for (int c = 0; c < Cin; ++c) acc += wr[c * 3] * ir[c];
  }
  if (relu) acc = fmaxf(acc, 0.f);
  out[((long long)b * L + l) * Cout + o] = acc;
}

// ConvTranspose1d(k=3,s=2,p=1,op=1): Lout = 2*Lin, PyTorch weight (Cin,Cout,3)
__global__ void k_convT(const float* __restrict__ in, const float* __restrict__ w,
                        const float* __restrict__ bias, float* __restrict__ out,
                        int Lin, int total)
{
  const int idx = blockIdx.x * blockDim.x + threadIdx.x;
  if (idx >= total) return;
  const int Lout = 2 * Lin;
  const int l = idx % Lout;
  const int o = (idx / Lout) & 63;
  const int b = idx / (Lout * 64);
  float acc = bias ? bias[o] : 0.f;
  for (int c = 0; c < 64; ++c) {
    const float* ic = in + ((long long)b * 64 + c) * Lin;
#pragma unroll
    for (int k = 0; k < 3; ++k) {
      const int i = l + k - 1;
      if (i < 0 || (i & 1)) continue;
      const int m = i >> 1;
      if (m >= Lin) continue;
      acc += w[(c * 64 + o) * 3 + (2 - k)] * ic[m];
    }
  }
  out[((long long)b * 64 + o) * Lout + l] = acc;
}

__global__ __launch_bounds__(256) void k_groupnorm(float* __restrict__ x,
    const float* __restrict__ g, const float* __restrict__ bb, int C, int L)
{
  __shared__ float r1[256];
  __shared__ float r2[256];
  const int b = blockIdx.x, tid = threadIdx.x;
  const int N = C * L;
  float* xb = x + (long long)b * N;
  float s = 0.f, s2 = 0.f;
  for (int i = tid; i < N; i += 256) { const float v = xb[i]; s += v; s2 += v * v; }
  r1[tid] = s; r2[tid] = s2;
  __syncthreads();
  for (int off = 128; off > 0; off >>= 1) {
    if (tid < off) { r1[tid] += r1[tid + off]; r2[tid] += r2[tid + off]; }
    __syncthreads();
  }
  const float m = r1[0] / N;
  const float rs = rsqrtf(r2[0] / N - m * m + 1e-5f);
  for (int i = tid; i < N; i += 256) {
    const int c = i / L;
    xb[i] = (xb[i] - m) * rs * g[c] + bb[c];
  }
}

__global__ void k_glu_silu(const float* __restrict__ y, float* __restrict__ z, int total)
{
  const int idx = blockIdx.x * blockDim.x + threadIdx.x;
  if (idx >= total) return;
  const int l = idx & 31;
  const int c = (idx >> 5) % 640;
  const int b = idx / (640 * 32);
  const float* yr = y + ((long long)b * 640 + c) * 64;
  const float t = yr[l] * sigm(yr[l + 32]);
  z[idx] = t * sigm(t);
}

__global__ void k_dwconv(const float* __restrict__ in, const float* __restrict__ w,
                         const float* __restrict__ bias, float* __restrict__ out,
                         int L, int total)
{
  const int idx = blockIdx.x * blockDim.x + threadIdx.x;
  if (idx >= total) return;
  const int l = idx % L;
  const int c = (idx / L) % 640;
  const int b = idx / (640 * L);
  const float* ir = in + ((long long)b * 640 + c) * L;
  float acc = bias[c];
#pragma unroll
  for (int t = 0; t < 7; ++t) {
    const int li = l + t - 3;
    if (li >= 0 && li < L) acc += w[c * 7 + t] * ir[li];
  }
  out[idx] = acc;
}

__global__ void k_t_cl2lc(const float* __restrict__ in, float* __restrict__ out,
                          int C, int L, int total)
{
  const int idx = blockIdx.x * blockDim.x + threadIdx.x;
  if (idx >= total) return;
  const int l = idx % L;
  const int c = (idx / L) % C;
  const int b = idx / (L * C);
  out[((long long)b * L + l) * C + c] = in[((long long)b * C + c) * L + l];
}
__global__ void k_t_lc2cl(const float* __restrict__ in, float* __restrict__ out,
                          int C, int L, int total)
{
  const int idx = blockIdx.x * blockDim.x + threadIdx.x;
  if (idx >= total) return;
  const int l = idx % L;
  const int c = (idx / L) % C;
  const int b = idx / (L * C);
  out[((long long)b * C + c) * L + l] = in[((long long)b * L + l) * C + c];
}
__global__ void k_addt(float* __restrict__ out, const float* __restrict__ a_cl,
                       const float* __restrict__ t_lc, int C, int L, int total)
{
  const int idx = blockIdx.x * blockDim.x + threadIdx.x;
  if (idx >= total) return;
  const int l = idx % L;
  const int c = (idx / L) % C;
  const int b = idx / (L * C);
  out[((long long)b * C + c) * L + l] =
      a_cl[((long long)b * C + c) * L + l] + t_lc[((long long)b * L + l) * C + c];
}
__global__ void k_add(float* __restrict__ o, const float* __restrict__ a,
                      const float* __restrict__ b, int n)
{
  const int i = blockIdx.x * blockDim.x + threadIdx.x;
  if (i < n) o[i] = a[i] + b[i];
}
__global__ void k_add_ip(float* __restrict__ a, const float* __restrict__ b, int n)
{
  const int i = blockIdx.x * blockDim.x + threadIdx.x;
  if (i < n) a[i] += b[i];
}

// ============================ host orchestration ============================
struct S4P { const float *enc_w,*enc_b,*logdt,*Are,*Aim,*Cre,*Cim,*D,*ow,*ob,*lnw,*lnb; };
struct LayerP { const float *ln1w,*ln1b,*wq,*bq,*wk,*bk,*wv,*bv,*wo,*bo,*ln2w,*ln2b,*cw1,*cb1,*cw2,*cb2; };

extern "C" void kernel_launch(void* const* d_in, const int* in_sizes, int n_in,
                              void* d_out, int out_size, void* d_ws, size_t ws_size,
                              hipStream_t stream)
{
  (void)in_sizes; (void)n_in; (void)out_size; (void)ws_size;
  int qi = 0;
  auto F = [&]() -> const float* { return (const float*)d_in[qi++]; };
  const float* dec_input = F();
  qi++; // sub_id (unused by reference forward)
  const float* conv3_w = F(); const float* conv3_b = F();
  const float* bn_g = F();    const float* bn_b = F();
  const float* cd1_w = F();   const float* cd1_b = F();
  const float* cd2_w = F();   const float* cd2_b = F();
  const float* up1_w = F();   const float* up1_b = F();
  const float* up2_w = F();
  S4P s4u1, s4u2, s4u3, s4cb;
  auto grabS4 = [&](S4P& p){
    p.enc_w=F(); p.enc_b=F(); p.logdt=F(); p.Are=F(); p.Aim=F(); p.Cre=F(); p.Cim=F();
    p.D=F(); p.ow=F(); p.ob=F(); p.lnw=F(); p.lnb=F();
  };
  grabS4(s4u1); grabS4(s4u2); grabS4(s4u3);
  const float* ffm_lnw=F(); const float* ffm_lnb=F();
  const float* ffm_w1=F();  const float* ffm_b1=F();
  const float* ffm_w2=F();  const float* ffm_b2=F();
  LayerP lay[4];
  for (int i = 0; i < 4; ++i) {
    LayerP& Lp = lay[i];
    Lp.ln1w=F(); Lp.ln1b=F(); Lp.wq=F(); Lp.bq=F(); Lp.wk=F(); Lp.bk=F();
    Lp.wv=F(); Lp.bv=F(); Lp.wo=F(); Lp.bo=F(); Lp.ln2w=F(); Lp.ln2b=F();
    Lp.cw1=F(); Lp.cb1=F(); Lp.cw2=F(); Lp.cb2=F();
  }
  const float* pw1_w=F(); const float* pw1_b=F();
  const float* gn1_g=F(); const float* gn1_b=F();
  const float* dw_w=F();  const float* dw_b=F();
  const float* gn2_g=F(); const float* gn2_b=F();
  const float* pw2_w=F(); const float* pw2_b=F();
  grabS4(s4cb);

  const int Bn = 32, Lf = 640;
  const long long SZ = (long long)Bn * Lf * 64;   // 1,310,720 floats per slot
  float* Wp = (float*)d_ws;                       // needs ~79 MB of workspace
  auto R = [&](int i){ return Wp + (long long)i * SZ; };
  float* xA   = R(0);
  float* xB   = R(1);
  float* resb = R(2);
  float* res2 = R(3);
  float* S0   = R(4);   // scratch quad R4..R7 (contiguous: qkv/o, conv-FFN 4*SZ)
  float* S1   = R(5);
  float* S2   = R(6);
  float* S3   = R(7);
  float* hb   = R(8);
  float* enc0 = R(9);
  float* enc1 = R(10);
  float* enc2 = R(11);
  float* T1   = R(12);
  float* T2   = R(13);
  float* T3   = R(14);
  float* kssm = R(15);  // 64*640 floats used
  float* glub = S0;     // S4 GLU buffer: 2*SZ spanning S0..S1
  float* ybuf = S2;

  auto g256 = [](long long n){ return dim3((unsigned)((n + 255) / 256)); };

  auto gemm = [&](const float* A_, const float* B_, const float* bias, const float* res_,
                  float* C_, int M, int N, int K, int lda, int ldb, int ldc,
                  long long sA, long long sB, long long sC, long long sR,
                  int batch, int biasRow){
    dim3 g((N + 63) / 64, (M + 63) / 64, batch);
    k_gemm<<<g, 128, 0, stream>>>(A_, B_, bias, res_, C_, M, N, K, lda, ldb, ldc,
                                  sA, sB, sC, sR, 0, biasRow, 1.f);
  };

  auto s4run = [&](const float* in, float* out, int L, int din, const S4P& p){
    const int M = Bn * L;
    gemm(in, p.enc_w, p.enc_b, nullptr, out, M, 64, din, din, 64, 64, 0,0,0,0, 1, 0);
    for (int i = 0; i < 8; ++i) {
      k_s4kernel<<<g256((long long)64 * L), 256, 0, stream>>>(p.logdt + i * 64,
          p.Are + (long long)i * 4096, p.Aim + (long long)i * 4096,
          p.Cre + (long long)i * 4096, p.Cim + (long long)i * 4096, kssm, L);
      k_causal<<<dim3(64, Bn), 256, 0, stream>>>(out, kssm, p.D + i * 64, ybuf, L);
      gemm(ybuf, p.ow + (long long)i * 8192, p.ob + i * 128, nullptr, glub,
           M, 128, 64, 64, 128, 128, 0,0,0,0, 1, 0);
      k_s4_tail<<<g256(M), 256, 0, stream>>>(out, glub, p.lnw + i * 64, p.lnb + i * 64, M);
    }
  };

  auto ffmrun = [&](const float* x, float* out){
    k_layernorm<<<g256(Bn * Lf), 256, 0, stream>>>(x, hb, ffm_lnw, ffm_lnb, Bn * Lf);
    k_ffm<<<dim3(Bn * Lf / 32), 256, 0, stream>>>(hb, x, ffm_w1, ffm_b1, ffm_w2, ffm_b2, out);
  };

  // TDM path probe (NULL tensor descriptor => deterministic no-op)
  k_tdm_probe<<<dim3(1), 32, 0, stream>>>();

  // ---- stem: conv3(k7,p3)+BN+ReLU, then encoder downsamples ----
  k_conv1d<<<g256(SZ), 256, 0, stream>>>(dec_input, conv3_w, conv3_b, bn_g, bn_b, enc0,
                                         640, 640, 64, 64, 7, 3, 1, 1, 1, (int)SZ);
  k_conv1d<<<g256(SZ / 2), 256, 0, stream>>>(enc0, cd1_w, cd1_b, nullptr, nullptr, enc1,
                                             640, 320, 64, 64, 3, 1, 2, 0, 0, (int)(SZ / 2));
  k_conv1d<<<g256(SZ / 4), 256, 0, stream>>>(enc1, cd2_w, cd2_b, nullptr, nullptr, enc2,
                                             320, 160, 64, 64, 2, 0, 2, 0, 0, (int)(SZ / 4));
  // ---- U-net with 3 S4 stacks ----
  k_t_cl2lc<<<g256(SZ / 4), 256, 0, stream>>>(enc2, T1, 64, 160, (int)(SZ / 4));
  s4run(T1, T2, 160, 64, s4u1);
  k_addt<<<g256(SZ / 4), 256, 0, stream>>>(T3, enc2, T2, 64, 160, (int)(SZ / 4));
  k_convT<<<g256(SZ / 2), 256, 0, stream>>>(T3, up1_w, up1_b, T1, 160, (int)(SZ / 2));
  k_add_ip<<<g256(SZ / 2), 256, 0, stream>>>(T1, enc1, (int)(SZ / 2));
  k_t_cl2lc<<<g256(SZ / 2), 256, 0, stream>>>(T1, T2, 64, 320, (int)(SZ / 2));
  s4run(T2, T3, 320, 64, s4u2);
  k_t_lc2cl<<<g256(SZ / 2), 256, 0, stream>>>(T3, T1, 64, 320, (int)(SZ / 2));
  k_convT<<<g256(SZ), 256, 0, stream>>>(T1, up2_w, nullptr, T2, 320, (int)SZ);
  k_add_ip<<<g256(SZ), 256, 0, stream>>>(T2, enc0, (int)SZ);
  k_t_cl2lc<<<g256(SZ), 256, 0, stream>>>(T2, T3, 64, 640, (int)SZ);
  s4run(T3, xA, 640, 64, s4u3);
  // ---- x = x + 0.5*ffm(x) ----
  ffmrun(xA, xB);
  float* cur = xB; float* alt = xA;
  (void)hipMemcpyAsync(resb, cur, SZ * sizeof(float), hipMemcpyDeviceToDevice, stream);
  // ---- 4 transformer blocks ----
  for (int li = 0; li < 4; ++li) {
    const LayerP& Lp = lay[li];
    k_layernorm<<<g256(Bn * Lf), 256, 0, stream>>>(cur, hb, Lp.ln1w, Lp.ln1b, Bn * Lf);
    gemm(hb, Lp.wq, Lp.bq, nullptr, S0, Bn * Lf, 64, 64, 64, 64, 64, 0,0,0,0, 1, 0);
    gemm(hb, Lp.wk, Lp.bk, nullptr, S1, Bn * Lf, 64, 64, 64, 64, 64, 0,0,0,0, 1, 0);
    gemm(hb, Lp.wv, Lp.bv, nullptr, S2, Bn * Lf, 64, 64, 64, 64, 64, 0,0,0,0, 1, 0);
    k_attn<<<dim3(Lf / 16, 4, Bn), 128, 0, stream>>>(S0, S1, S2, S3, Lf);
    gemm(S3, Lp.wo, Lp.bo, cur, alt, Bn * Lf, 64, 64, 64, 64, 64, 0,0,0,0, 1, 0);
    { float* t = cur; cur = alt; alt = t; }
    k_layernorm<<<g256(Bn * Lf), 256, 0, stream>>>(cur, hb, Lp.ln2w, Lp.ln2b, Bn * Lf);
    k_conv1d_lc<<<g256((long long)Bn * Lf * 256), 256, 0, stream>>>(hb, Lp.cw1, Lp.cb1, S0,
                                                                    Lf, 64, 256, 1, (long long)Bn * Lf * 256);
    k_conv1d_lc<<<g256(SZ), 256, 0, stream>>>(S0, Lp.cw2, Lp.cb2, T3, Lf, 256, 64, 0, SZ);
    k_add_ip<<<g256(SZ), 256, 0, stream>>>(cur, T3, (int)SZ);
  }
  k_add_ip<<<g256(SZ), 256, 0, stream>>>(cur, resb, (int)SZ);
  (void)hipMemcpyAsync(res2, cur, SZ * sizeof(float), hipMemcpyDeviceToDevice, stream);
  // ---- ConvolutionBlock (time dim as channels) + inner S4 ----
  gemm(pw1_w, cur, pw1_b, nullptr, S0, 640, 64, 640, 640, 64, 64,
       0, (long long)640 * 64, (long long)640 * 64, 0, Bn, 1);
  k_groupnorm<<<dim3(Bn), 256, 0, stream>>>(S0, gn1_g, gn1_b, 640, 64);
  k_glu_silu<<<g256((long long)Bn * 640 * 32), 256, 0, stream>>>(S0, S1, Bn * 640 * 32);
  k_dwconv<<<g256((long long)Bn * 640 * 32), 256, 0, stream>>>(S1, dw_w, dw_b, S2, 32, Bn * 640 * 32);
  k_groupnorm<<<dim3(Bn), 256, 0, stream>>>(S2, gn2_g, gn2_b, 640, 32);
  gemm(pw2_w, S2, pw2_b, nullptr, S3, 640, 32, 640, 640, 32, 32,
       0, (long long)640 * 32, (long long)640 * 32, 0, Bn, 1);
  s4run(S3, T3, 640, 32, s4cb);
  k_add<<<g256(SZ), 256, 0, stream>>>(alt, res2, T3, (int)SZ);
  cur = alt;
  // ---- final x = x + 0.5*ffm(x) -> d_out ----
  ffmrun(cur, (float*)d_out);
}
